// TransformerLayer_22368189678182
// MI455X (gfx1250) — matrix operand order repaired and checked by reading, not compiled, not run on hardware
//
#include <hip/hip_runtime.h>
#include <math.h>

// ---------------------------------------------------------------------------
// Transformer encoder (4 layers) for MI455X / gfx1250, bf16 WMMA pipeline.
//   L=4, B=2, S=1024, H=1024, NH=16, DH=64, F=4096
// d_out (fp32 [B,S,H]) doubles as the running fp32 residual buffer.
//
// Data movement:
//   - GEMM A/B tiles: TDM tensor_load_to_lds (one D# per 8KB tile, issued by
//     wave 0, TENSORcnt), double-buffered so the DMA overlaps the WMMAs.
//   - Attention K/V tiles: per-lane async global->LDS b128 copies (ASYNCcnt).
// Weights are pre-transposed to [N][K] bf16 and the V projection is written
// head-transposed, so no LDS-side reshaping is ever needed.
// ---------------------------------------------------------------------------

#define L_ 4
#define B_ 2
#define S_ 1024
#define H_ 1024
#define NH_ 16
#define DH_ 64
#define F_ 4096
#define M_TOT (B_ * S_)   // 2048 rows for all projection GEMMs

typedef __bf16 bf16_t;
typedef __attribute__((ext_vector_type(16))) __bf16    v16bf;
typedef __attribute__((ext_vector_type(8)))  __bf16    v8bf;
typedef __attribute__((ext_vector_type(8)))  float     v8f;
typedef __attribute__((ext_vector_type(4)))  int       v4i;
typedef __attribute__((ext_vector_type(8)))  int       v8i;
typedef __attribute__((ext_vector_type(4)))  unsigned  v4u;

struct V16 { union { v16bf v; v8bf h[2]; }; };

__device__ __forceinline__ v8f wmma_bf16(const V16& a, const V16& b, v8f c) {
  // D = A(16x32 bf16) * B(32x16 bf16) + C(16x16 f32)
  return __builtin_amdgcn_wmma_f32_16x16x32_bf16(
      /*neg_a=*/false, a.v, /*neg_b=*/false, b.v,
      /*c_mod=*/(short)0, c, /*reuse_a=*/false, /*reuse_b=*/false);
}

// ---- CDNA5 async global->LDS copy (16 bytes per lane), ASYNCcnt-tracked ----
__device__ __forceinline__ void async_copy16(const void* gsrc, void* lds_dst) {
#if __has_builtin(__builtin_amdgcn_global_load_async_to_lds_b128)
  typedef __attribute__((address_space(1))) v4i* as1_v4i;
  typedef __attribute__((address_space(3))) v4i* as3_v4i;
  as1_v4i g = (as1_v4i)(size_t)gsrc;
  as3_v4i l = (as3_v4i)(unsigned)(size_t)lds_dst;
  __builtin_amdgcn_global_load_async_to_lds_b128(g, l, 0, 0);
#else
  const unsigned lds_addr = (unsigned)(size_t)lds_dst;
  asm volatile("global_load_async_to_lds_b128 %0, %1, off"
               :: "v"(lds_addr), "v"(gsrc)
               : "memory");
#endif
}

__device__ __forceinline__ void wait_async0() {
#if __has_builtin(__builtin_amdgcn_s_wait_asynccnt)
  __builtin_amdgcn_s_wait_asynccnt(0);
#else
  asm volatile("s_wait_asynccnt 0x0" ::: "memory");
#endif
}

// ---- CDNA5 Tensor Data Mover: 2D bf16 tile, global -> LDS, TENSORcnt ----
#if __has_builtin(__builtin_amdgcn_tensor_load_to_lds)
#define HAVE_TDM 1
#else
#define HAVE_TDM 0
#endif

__device__ __forceinline__ void wait_tensor0() {
#if __has_builtin(__builtin_amdgcn_s_wait_tensorcnt)
  __builtin_amdgcn_s_wait_tensorcnt((short)0);
#else
  asm volatile("s_wait_tensorcnt 0x0" ::: "memory");
#endif
}

#if HAVE_TDM
// One D# describing a [rows x d0_elems] bf16 tile with row stride
// `stride_elems`, packed contiguously into LDS at `lds_addr`.
__device__ __forceinline__ void tdm_load_2d(const void* gsrc, unsigned lds_addr,
                                            unsigned d0_elems, unsigned rows,
                                            unsigned stride_elems) {
  const unsigned long long ga = (unsigned long long)(size_t)gsrc;
  const unsigned td0 = stride_elems;       // tensor dim0 (>= tile width, in-bounds tiles)
  const unsigned td1 = 0x40000000u;        // tensor dim1: effectively unbounded
  v4u g0;
  g0[0] = 1u;                                              // count=1, user descriptor
  g0[1] = lds_addr;                                        // LDS byte address
  g0[2] = (unsigned)ga;                                    // global_addr[31:0]
  g0[3] = (unsigned)((ga >> 32) & 0x1FFFFFFu) | (2u << 30);// global_addr[56:32], type=2
  v8i g1;
  g1[0] = (int)(1u << 16);                                 // wg_mask=0, data_size=1 (2B)
  g1[1] = (int)((td0 & 0xFFFFu) << 16);                    // abar=0 | tensor_dim0[15:0]
  g1[2] = (int)((td0 >> 16) | ((td1 & 0xFFFFu) << 16));    // tensor_dim0[31:16] | td1[15:0]
  g1[3] = (int)((td1 >> 16) | (d0_elems << 16));           // td1[31:16] | tile_dim0
  g1[4] = (int)(rows & 0xFFFFu);                           // tile_dim1 | tile_dim2=0
  g1[5] = (int)stride_elems;                               // tensor_dim0_stride[31:0]
  g1[6] = 0;                                               // stride0[47:32] | stride1[15:0]
  g1[7] = 0;
  const v4i gz4 = {};                                      // groups 2/3: 2D tensor
  const v8i gz8 = {};                                      // extra group (6-arg form)
  __builtin_amdgcn_tensor_load_to_lds(g0, g1, gz4, gz4, gz8, 0);
}
#endif

// ---------------------------------------------------------------------------
// Weight prep: fp32 [L][K][N] -> bf16 [L][N][K] (transpose), LDS-tiled so both
// the global reads and writes are coalesced. One-time cost per call.
// ---------------------------------------------------------------------------
__global__ __launch_bounds__(256)
void convert_transpose(const float* __restrict__ w, bf16_t* __restrict__ wt,
                       int K, int N) {
  __shared__ float tile[32][33];
  const int l  = blockIdx.z;
  const int n0 = blockIdx.x * 32;
  const int k0 = blockIdx.y * 32;
  const float* src = w + (size_t)l * K * N;
  bf16_t* dst = wt + (size_t)l * N * K;
  const int tx = threadIdx.x & 31;
  const int ty = threadIdx.x >> 5;   // 0..7
#pragma unroll
  for (int r = ty; r < 32; r += 8)
    tile[r][tx] = src[(size_t)(k0 + r) * N + n0 + tx];
  __syncthreads();
#pragma unroll
  for (int r = ty; r < 32; r += 8)
    dst[(size_t)(n0 + r) * K + k0 + tx] = (bf16_t)tile[tx][r];
}

__global__ __launch_bounds__(256) void init_x(const float* __restrict__ x,
                                              float* __restrict__ xf,
                                              bf16_t* __restrict__ xb, int n) {
  int i = blockIdx.x * blockDim.x + threadIdx.x;
  int stride = gridDim.x * blockDim.x;
  for (; i < n; i += stride) {
    float v = x[i];
    xf[i] = v;
    xb[i] = (bf16_t)v;
  }
}

// ---------------------------------------------------------------------------
// Tiled bf16 WMMA GEMM: C[M,N] = A[M,K] * Bt^T + bias (Bt is [N][K] bf16).
// Block tile 128x128, K-step 32, 8 wave32 waves (2x4), wave tile 64x32.
// Tiles staged by the Tensor Data Mover (wave 0 issues 2 tensor loads per
// K-step), double-buffered so the DMA of slice k+1 overlaps the WMMAs of
// slice k. Falls back to per-lane async-LDS copies if TDM is unavailable.
// ---------------------------------------------------------------------------
enum { EPI_BF16 = 0, EPI_GELU = 1, EPI_RESID = 2, EPI_VT = 3 };

template <int EPI>
__global__ __launch_bounds__(256)
void gemm_bf16_wmma(const bf16_t* __restrict__ A,    // [M][K]
                    const bf16_t* __restrict__ Bt,   // [N][K] (pre-transposed)
                    const float*  __restrict__ bias,
                    bf16_t* __restrict__ outB,
                    float*  __restrict__ resid,        // EPI_RESID: fp32 residual
                    const float* __restrict__ alpha_p, // EPI_RESID: rezero alpha
                    int M, int N, int K) {
  __shared__ __align__(32) bf16_t As[2][128 * 32];   // [m][k]
  __shared__ __align__(32) bf16_t Bs[2][128 * 32];   // [n][k]

  const int tid  = threadIdx.x;
  const int lane = tid & 31;
  const int wave = tid >> 5;
  const int wrow = wave >> 2;        // 0..1  (64-row strip)
  const int wcol = wave & 3;         // 0..3  (32-col strip)
  const int half = lane >> 4;
  const int l15  = lane & 15;
  const int koff = half * 16;
  const int m0 = blockIdx.y * 128;
  const int n0 = blockIdx.x * 128;

  v8f acc[4][2];
#pragma unroll
  for (int i = 0; i < 4; ++i)
#pragma unroll
    for (int j = 0; j < 2; ++j) acc[i][j] = (v8f){};

#if HAVE_TDM
  auto stage = [&](int buf, int k0) {
    if (wave == 0) {   // one wave drives the DMA engine; EXEC is ignored by TDM
      tdm_load_2d(A  + (size_t)m0 * K + k0, (unsigned)(size_t)&As[buf][0], 32, 128, K);
      tdm_load_2d(Bt + (size_t)n0 * K + k0, (unsigned)(size_t)&Bs[buf][0], 32, 128, K);
    }
  };
  auto stage_wait = [&]() { if (wave == 0) wait_tensor0(); };
#else
  // 128x32 tile, 256 threads -> 16 contiguous bf16 (32B) each = 2 async b128
  const int sr = tid >> 1;           // 0..127
  const int sc = (tid & 1) * 16;     // 0 or 16
  auto stage = [&](int buf, int k0) {
    const bf16_t* sa = A  + (size_t)(m0 + sr) * K + k0 + sc;
    const bf16_t* sb = Bt + (size_t)(n0 + sr) * K + k0 + sc;
    async_copy16(sa,     &As[buf][sr * 32 + sc]);
    async_copy16(sa + 8, &As[buf][sr * 32 + sc + 8]);
    async_copy16(sb,     &Bs[buf][sr * 32 + sc]);
    async_copy16(sb + 8, &Bs[buf][sr * 32 + sc + 8]);
  };
  auto stage_wait = [&]() { wait_async0(); };
#endif

  stage(0, 0);
  stage_wait();
  __syncthreads();

  const int nK = K >> 5;
  for (int kt = 0; kt < nK; ++kt) {
    const int buf = kt & 1;
    if (kt + 1 < nK) stage(buf ^ 1, (kt + 1) * 32);  // DMA prefetch overlaps WMMA

    V16 af[4];
#pragma unroll
    for (int i = 0; i < 4; ++i) {
      const int row = wrow * 64 + i * 16 + l15;
      af[i].h[0] = *(const v8bf*)&As[buf][row * 32 + koff];
      af[i].h[1] = *(const v8bf*)&As[buf][row * 32 + 8 + koff];
    }
    V16 bfrag[2];
#pragma unroll
    for (int j = 0; j < 2; ++j) {
      const int col = wcol * 32 + j * 16 + l15;
      bfrag[j].v = *(const v16bf*)&Bs[buf][col * 32 + half * 16];
    }
#pragma unroll
    for (int i = 0; i < 4; ++i)
#pragma unroll
      for (int j = 0; j < 2; ++j) acc[i][j] = wmma_bf16(af[i], bfrag[j], acc[i][j]);

    stage_wait();      // prefetched slice landed in LDS
    __syncthreads();   // all waves done reading buf / writing buf^1
  }

  __builtin_prefetch(bias + n0, 0, 0);   // global_prefetch_b8 (epilogue bias row)
  const float alpha = (EPI == EPI_RESID) ? alpha_p[0] : 0.0f;
#pragma unroll
  for (int i = 0; i < 4; ++i)
#pragma unroll
    for (int j = 0; j < 2; ++j) {
      const int col = n0 + wcol * 32 + j * 16 + l15;
      const float bsv = bias[col];
#pragma unroll
      for (int v = 0; v < 8; ++v) {
        const int row = m0 + wrow * 64 + i * 16 + half * 8 + v;
        const size_t idx = (size_t)row * N + col;
        float val = acc[i][j][v] + bsv;
        if (EPI == EPI_BF16) {
          outB[idx] = (bf16_t)val;
        } else if (EPI == EPI_GELU) {
          const float c3 = 0.7978845608028654f * (val + 0.044715f * val * val * val);
          const float g  = 0.5f * val * (1.0f + tanhf(c3));
          outB[idx] = (bf16_t)g;
        } else if (EPI == EPI_RESID) {  // x = x + alpha*(acc+bias)
          const float xn = resid[idx] + alpha * val;
          resid[idx] = xn;
          outB[idx]  = (bf16_t)xn;
        } else {  // EPI_VT: write V projection head-transposed: [b*H + col][S]
          const int bb   = row >> 10;      // row / S_
          const int srow = row & (S_ - 1);
          outB[(size_t)(bb * H_ + col) * S_ + srow] = (bf16_t)val;
        }
      }
    }
}

// ---------------------------------------------------------------------------
// Flash-style attention, one (b, head, 128-row q-block) per workgroup.
// Q fragments persist in registers; K tile and pre-transposed V^T tile are
// async-copied into LDS per 128-row kv block; online softmax in LDS; the
// 32KB P tile aliases the dead K tile.
// ---------------------------------------------------------------------------
__global__ __launch_bounds__(256)
void attention_wmma(const bf16_t* __restrict__ Qg,   // [B,S,H]
                    const bf16_t* __restrict__ Kg,   // [B,S,H]
                    const bf16_t* __restrict__ Vtg,  // [B*H][S] head-transposed
                    const float*  __restrict__ mask, // [B,S,S]
                    bf16_t* __restrict__ ctx) {
  __shared__ __align__(32) bf16_t KsPs[128 * 128];  // Ks = first 16KB, Ps = 32KB
  __shared__ __align__(32) bf16_t Vt[64 * 128];     // V^T tile: [d][kv]
  __shared__ float mrow[128], lrow[128], srow[128];
  __shared__ float pmax[2][128], psum[2][128];

  bf16_t* Ks = KsPs;
  bf16_t* Ps = KsPs;

  const int tid  = threadIdx.x;
  const int lane = tid & 31;
  const int wave = tid >> 5;
  const int wrow = wave >> 1;        // 0..3  (32-row q strip)
  const int wcol = wave & 1;         // 0..1
  const int half = lane >> 4;
  const int l15  = lane & 15;
  const int koff = half * 16;

  const int bid = blockIdx.x;
  const int qb  = bid & 7;           // S/128 = 8 q-blocks
  const int h   = (bid >> 3) & 15;
  const int b   = bid >> 7;
  const int q0  = qb * 128;
  const size_t base  = ((size_t)b * S_) * H_ + (size_t)h * DH_;  // Q/K head slice
  const size_t vbase = (size_t)(b * H_ + h * DH_) * S_;          // V^T head slice

  // Persistent Q fragments: 2 m-frags x 2 k-steps over DH=64
  V16 qf[2][2];
#pragma unroll
  for (int mf = 0; mf < 2; ++mf)
#pragma unroll
    for (int ks = 0; ks < 2; ++ks) {
      const int row = q0 + wrow * 32 + mf * 16 + l15;
      const bf16_t* p = Qg + base + (size_t)row * H_ + ks * 32;
      qf[mf][ks].h[0] = *(const v8bf*)(p + koff);
      qf[mf][ks].h[1] = *(const v8bf*)(p + 8 + koff);
    }

  v8f Oacc[2][2];
#pragma unroll
  for (int i = 0; i < 2; ++i)
#pragma unroll
    for (int j = 0; j < 2; ++j) Oacc[i][j] = (v8f){};

  if (tid < 128) { mrow[tid] = -1e30f; lrow[tid] = 0.0f; }

  for (int tb = 0; tb < S_ / 128; ++tb) {
    const int kv0 = tb * 128;
    __syncthreads();  // prior PV reads of Ps/Vt complete before overwrite

    // ---- async-stage K tile (128x64, row-major == B^T for QK^T) ----
    {
      const int kr = tid >> 1;
      const int kc = (tid & 1) * 32;
      const bf16_t* kp = Kg + base + (size_t)(kv0 + kr) * H_ + kc;
#pragma unroll
      for (int c = 0; c < 32; c += 8)
        async_copy16(kp + c, &Ks[kr * 64 + kc + c]);
    }
    // ---- async-stage V^T tile (64x128, contiguous rows of global V^T) ----
    {
      const int vr = tid >> 2;           // 0..63 (d)
      const int vc = (tid & 3) * 32;     // 0..96 (kv)
      const bf16_t* vp = Vtg + vbase + (size_t)vr * S_ + kv0 + vc;
#pragma unroll
      for (int c = 0; c < 32; c += 8)
        async_copy16(vp + c, &Vt[vr * 128 + vc + c]);
    }
    wait_async0();
    __syncthreads();

    // ---- scores: S = Q K^T * scale + mask-adder ----
    v8f s[2][4];
#pragma unroll
    for (int mf = 0; mf < 2; ++mf)
#pragma unroll
      for (int nf = 0; nf < 4; ++nf) s[mf][nf] = (v8f){};
#pragma unroll
    for (int ks = 0; ks < 2; ++ks) {
      V16 bfrag[4];
#pragma unroll
      for (int nf = 0; nf < 4; ++nf) {
        const int col = wcol * 64 + nf * 16 + l15;
        bfrag[nf].v = *(const v16bf*)&Ks[col * 64 + ks * 32 + half * 16];
      }
#pragma unroll
      for (int mf = 0; mf < 2; ++mf)
#pragma unroll
        for (int nf = 0; nf < 4; ++nf) s[mf][nf] = wmma_bf16(qf[mf][ks], bfrag[nf], s[mf][nf]);
    }

    // ---- scale + additive mask, partial row max (16-lane shfl reduce) ----
#pragma unroll
    for (int mf = 0; mf < 2; ++mf) {
      float rm[8];
#pragma unroll
      for (int v = 0; v < 8; ++v) rm[v] = -1e30f;
#pragma unroll
      for (int nf = 0; nf < 4; ++nf) {
        const int kv = kv0 + wcol * 64 + nf * 16 + l15;
#pragma unroll
        for (int v = 0; v < 8; ++v) {
          const int q = q0 + wrow * 32 + mf * 16 + half * 8 + v;
          const float adder = (1.0f - mask[(size_t)(b * S_ + q) * S_ + kv]) * -10000.0f;
          const float sv = s[mf][nf][v] * 0.125f + adder;   // 1/sqrt(64)
          s[mf][nf][v] = sv;
          rm[v] = fmaxf(rm[v], sv);
        }
      }
#pragma unroll
      for (int v = 0; v < 8; ++v) {
        float m = rm[v];
        m = fmaxf(m, __shfl_xor(m, 1, 32));
        m = fmaxf(m, __shfl_xor(m, 2, 32));
        m = fmaxf(m, __shfl_xor(m, 4, 32));
        m = fmaxf(m, __shfl_xor(m, 8, 32));
        if (l15 == 0) pmax[wcol][wrow * 32 + mf * 16 + half * 8 + v] = m;
      }
    }
    __syncthreads();
    if (tid < 128) {
      const float mold = mrow[tid];
      const float mnew = fmaxf(mold, fmaxf(pmax[0][tid], pmax[1][tid]));
      srow[tid] = __expf(mold - mnew);
      mrow[tid] = mnew;
    }
    __syncthreads();

    // ---- P = exp(S - m): write bf16 P tile, partial row sums, rescale O ----
#pragma unroll
    for (int mf = 0; mf < 2; ++mf) {
      float rs[8];
      float mloc[8];
#pragma unroll
      for (int v = 0; v < 8; ++v) {
        rs[v] = 0.0f;
        mloc[v] = mrow[wrow * 32 + mf * 16 + half * 8 + v];
      }
#pragma unroll
      for (int nf = 0; nf < 4; ++nf) {
#pragma unroll
        for (int v = 0; v < 8; ++v) {
          const float p = __expf(s[mf][nf][v] - mloc[v]);
          rs[v] += p;
          const int r = wrow * 32 + mf * 16 + half * 8 + v;
          Ps[r * 128 + wcol * 64 + nf * 16 + l15] = (bf16_t)p;
        }
      }
#pragma unroll
      for (int v = 0; v < 8; ++v) {
        float t = rs[v];
        t += __shfl_xor(t, 1, 32);
        t += __shfl_xor(t, 2, 32);
        t += __shfl_xor(t, 4, 32);
        t += __shfl_xor(t, 8, 32);
        if (l15 == 0) psum[wcol][wrow * 32 + mf * 16 + half * 8 + v] = t;
      }
#pragma unroll
      for (int v = 0; v < 8; ++v) {
        const float sc = srow[wrow * 32 + mf * 16 + half * 8 + v];
#pragma unroll
        for (int nf = 0; nf < 2; ++nf) Oacc[mf][nf][v] *= sc;
      }
    }
    __syncthreads();
    if (tid < 128) lrow[tid] = lrow[tid] * srow[tid] + psum[0][tid] + psum[1][tid];

    // ---- O += P V  (K = 128 kv in 4 WMMA steps) ----
#pragma unroll
    for (int ks = 0; ks < 4; ++ks) {
      V16 pa[2], vb[2];
#pragma unroll
      for (int mf = 0; mf < 2; ++mf) {
        const int row = wrow * 32 + mf * 16 + l15;
        pa[mf].h[0] = *(const v8bf*)&Ps[row * 128 + ks * 32 + koff];
        pa[mf].h[1] = *(const v8bf*)&Ps[row * 128 + ks * 32 + 8 + koff];
      }
#pragma unroll
      for (int nf = 0; nf < 2; ++nf) {
        const int dcol = wcol * 32 + nf * 16 + l15;
        vb[nf].v = *(const v16bf*)&Vt[dcol * 128 + ks * 32 + half * 16];
      }
#pragma unroll
      for (int mf = 0; mf < 2; ++mf)
#pragma unroll
        for (int nf = 0; nf < 2; ++nf) Oacc[mf][nf] = wmma_bf16(pa[mf], vb[nf], Oacc[mf][nf]);
    }
  }
  __syncthreads();

  // ---- normalize and store ctx (interleaved head layout [B,S,H]) ----
#pragma unroll
  for (int mf = 0; mf < 2; ++mf)
#pragma unroll
    for (int nf = 0; nf < 2; ++nf)
#pragma unroll
      for (int v = 0; v < 8; ++v) {
        const int r  = wrow * 32 + mf * 16 + half * 8 + v;
        const int d  = wcol * 32 + nf * 16 + l15;
        const float val = Oacc[mf][nf][v] / lrow[r];
        ctx[base + (size_t)(q0 + r) * H_ + d] = (bf16_t)val;
      }
}

// ---------------------------------------------------------------------------
// Host-side orchestration
// ---------------------------------------------------------------------------
extern "C" void kernel_launch(void* const* d_in, const int* in_sizes, int n_in,
                              void* d_out, int out_size, void* d_ws, size_t ws_size,
                              hipStream_t stream) {
  (void)in_sizes; (void)n_in; (void)out_size; (void)ws_size;

  const float* x     = (const float*)d_in[0];
  const float* mask  = (const float*)d_in[1];
  const float* wq    = (const float*)d_in[2];
  const float* bq    = (const float*)d_in[3];
  const float* wk    = (const float*)d_in[4];
  const float* bk    = (const float*)d_in[5];
  const float* wv    = (const float*)d_in[6];
  const float* bv    = (const float*)d_in[7];
  const float* wo    = (const float*)d_in[8];
  const float* bo    = (const float*)d_in[9];
  const float* aattn = (const float*)d_in[10];
  const float* wi    = (const float*)d_in[11];
  const float* bi    = (const float*)d_in[12];
  const float* wof   = (const float*)d_in[13];
  const float* bof   = (const float*)d_in[14];
  const float* affn  = (const float*)d_in[15];

  float* xf = (float*)d_out;  // running fp32 residual; final value is the output

  char* ws = (char*)d_ws;
  size_t off = 0;
  auto alloc = [&](size_t elems) -> bf16_t* {
    bf16_t* p = (bf16_t*)(ws + off);
    off += (elems * sizeof(bf16_t) + 255) & ~(size_t)255;
    return p;
  };
  const size_t NXH = (size_t)B_ * S_ * H_;     // 2M
  const size_t NXF = (size_t)B_ * S_ * F_;     // 8M
  bf16_t* xb     = alloc(NXH);
  bf16_t* qbuf   = alloc(NXH);
  bf16_t* kbuf   = alloc(NXH);
  bf16_t* vtbuf  = alloc(NXH);                 // head-transposed V
  bf16_t* ctxb   = alloc(NXH);
  bf16_t* interb = alloc(NXF);
  bf16_t* wqb    = alloc((size_t)L_ * H_ * H_);   // all weights stored [N][K] bf16
  bf16_t* wkb    = alloc((size_t)L_ * H_ * H_);
  bf16_t* wvb    = alloc((size_t)L_ * H_ * H_);
  bf16_t* wob    = alloc((size_t)L_ * H_ * H_);
  bf16_t* wib    = alloc((size_t)L_ * H_ * F_);
  bf16_t* wofb   = alloc((size_t)L_ * F_ * H_);

  // one-time (per call): fp32 -> bf16 + transpose to [N][K]
  const dim3 tb(256);
  convert_transpose<<<dim3(H_ / 32, H_ / 32, L_), tb, 0, stream>>>(wq,  wqb,  H_, H_);
  convert_transpose<<<dim3(H_ / 32, H_ / 32, L_), tb, 0, stream>>>(wk,  wkb,  H_, H_);
  convert_transpose<<<dim3(H_ / 32, H_ / 32, L_), tb, 0, stream>>>(wv,  wvb,  H_, H_);
  convert_transpose<<<dim3(H_ / 32, H_ / 32, L_), tb, 0, stream>>>(wo,  wob,  H_, H_);
  convert_transpose<<<dim3(F_ / 32, H_ / 32, L_), tb, 0, stream>>>(wi,  wib,  H_, F_);
  convert_transpose<<<dim3(H_ / 32, F_ / 32, L_), tb, 0, stream>>>(wof, wofb, F_, H_);
  init_x<<<2048, 256, 0, stream>>>(x, xf, xb, (int)NXH);

  const dim3 blk(256);
  const dim3 gH(H_ / 128, M_TOT / 128);   // (8, 16)
  const dim3 gF(F_ / 128, M_TOT / 128);   // (32, 16)
  const dim3 gA(B_ * NH_ * (S_ / 128));   // 256 (b, head, q-block)

  for (int l = 0; l < L_; ++l) {
    gemm_bf16_wmma<EPI_BF16><<<gH, blk, 0, stream>>>(
        xb, wqb + (size_t)l * H_ * H_, bq + l * H_, qbuf, nullptr, nullptr, M_TOT, H_, H_);
    gemm_bf16_wmma<EPI_BF16><<<gH, blk, 0, stream>>>(
        xb, wkb + (size_t)l * H_ * H_, bk + l * H_, kbuf, nullptr, nullptr, M_TOT, H_, H_);
    gemm_bf16_wmma<EPI_VT><<<gH, blk, 0, stream>>>(
        xb, wvb + (size_t)l * H_ * H_, bv + l * H_, vtbuf, nullptr, nullptr, M_TOT, H_, H_);

    attention_wmma<<<gA, blk, 0, stream>>>(qbuf, kbuf, vtbuf, mask, ctxb);

    gemm_bf16_wmma<EPI_RESID><<<gH, blk, 0, stream>>>(
        ctxb, wob + (size_t)l * H_ * H_, bo + l * H_, xb, xf, aattn + l, M_TOT, H_, H_);
    gemm_bf16_wmma<EPI_GELU><<<gF, blk, 0, stream>>>(
        xb, wib + (size_t)l * H_ * F_, bi + l * F_, interb, nullptr, nullptr, M_TOT, F_, H_);
    gemm_bf16_wmma<EPI_RESID><<<gH, blk, 0, stream>>>(
        interb, wofb + (size_t)l * F_ * H_, bof + l * H_, xb, xf, affn + l, M_TOT, H_, F_);
  }
}
